// BiLSTMCell_38792144618011
// MI455X (gfx1250) — matrix operand order repaired and checked by reading, not compiled, not run on hardware
//
#include <hip/hip_runtime.h>

// ---------------------------------------------------------------------------
// BiLSTM-style fused cell for MI455X (gfx1250, wave32, WMMA).
//   g[b,gate,h] = x@Wx^T + hidden@Wh^T + neighbors@Wn^T + bias
//   next_cell   = sigm(g_f)*cell + sigm(g_i)*tanh(g_s)
//   next_hidden = sigm(g_o)*tanh(next_cell)
// Strategy: pack [x|hidden] and [Wx|Wh] to bf16, one K=2048 GEMM using
// v_wmma_f32_16x16x32_bf16; BM=128 tile, 8 WMMA/wave/K-step, double-buffered
// LDS with async-to-LDS loads (ASYNCcnt) when the builtin is available.
// ---------------------------------------------------------------------------

typedef __bf16 bf16_t;
typedef __attribute__((ext_vector_type(8)))  bf16_t bhalf8;   // 16 bytes
typedef __attribute__((ext_vector_type(16))) bf16_t v16bf;    // WMMA A/B operand
typedef __attribute__((ext_vector_type(8)))  float  v8f;      // WMMA C/D operand
typedef int v4i __attribute__((vector_size(16)));             // b128 unit

#define Bsz   8192
#define Hsz   1024
#define INsz  1024
#define Kc    2048      // IN + H concatenated K
#define Ngate 4

#define BM 128
#define BN 32
#define BK 32
#define NSTEP (Kc / BK)

#if defined(__HIP_DEVICE_COMPILE__) && __has_builtin(__builtin_amdgcn_global_load_async_to_lds_b128)
#define USE_ASYNC_LDS 1
#else
#define USE_ASYNC_LDS 0
#endif

// ---- pack [x | hidden] rows into bf16 x_cat[B][2048] ----------------------
__global__ void pack_xcat(const float* __restrict__ x,
                          const float* __restrict__ h,
                          bf16_t* __restrict__ xcat) {
    int tid = blockIdx.x * blockDim.x + threadIdx.x;   // B*Kc/8 threads
    int idx = tid << 3;                                // 8 elems per thread
    int row = idx >> 11;                               // / 2048
    int col = idx & (Kc - 1);
    const float* src = (col < INsz) ? (x + (size_t)row * INsz + col)
                                    : (h + (size_t)row * Hsz + (col - INsz));
    float4 f0 = ((const float4*)src)[0];
    float4 f1 = ((const float4*)src)[1];
    bhalf8 o;
    o[0] = (bf16_t)f0.x; o[1] = (bf16_t)f0.y; o[2] = (bf16_t)f0.z; o[3] = (bf16_t)f0.w;
    o[4] = (bf16_t)f1.x; o[5] = (bf16_t)f1.y; o[6] = (bf16_t)f1.z; o[7] = (bf16_t)f1.w;
    *(bhalf8*)(xcat + idx) = o;
}

// ---- pack [Wx | Wh] rows into bf16 w_cat[4*H][2048] -----------------------
__global__ void pack_wcat(const float* __restrict__ wx,   // [4*H, IN]
                          const float* __restrict__ wh,   // [4*H, H]
                          bf16_t* __restrict__ wcat) {
    int tid = blockIdx.x * blockDim.x + threadIdx.x;   // 4H*Kc/8 threads
    int idx = tid << 3;
    int row = idx >> 11;
    int col = idx & (Kc - 1);
    const float* src = (col < INsz) ? (wx + (size_t)row * INsz + col)
                                    : (wh + (size_t)row * Hsz + (col - INsz));
    float4 f0 = ((const float4*)src)[0];
    float4 f1 = ((const float4*)src)[1];
    bhalf8 o;
    o[0] = (bf16_t)f0.x; o[1] = (bf16_t)f0.y; o[2] = (bf16_t)f0.z; o[3] = (bf16_t)f0.w;
    o[4] = (bf16_t)f1.x; o[5] = (bf16_t)f1.y; o[6] = (bf16_t)f1.z; o[7] = (bf16_t)f1.w;
    *(bhalf8*)(wcat + idx) = o;
}

__device__ __forceinline__ float sigm(float v) {
    return 1.0f / (1.0f + __expf(-v));
}

#if USE_ASYNC_LDS
__device__ __forceinline__ void async_b128(const bf16_t* g, bf16_t* l) {
    __builtin_amdgcn_global_load_async_to_lds_b128(
        (__attribute__((address_space(1))) v4i*)g,
        (__attribute__((address_space(3))) v4i*)l,
        0, 0);
}
__device__ __forceinline__ void wait_async0() {
#if __has_builtin(__builtin_amdgcn_s_wait_asynccnt)
    __builtin_amdgcn_s_wait_asynccnt(0);
#else
    asm volatile("s_wait_asynccnt 0x0" ::: "memory");
#endif
}
#endif

// ---- main fused GEMM + LSTM epilogue --------------------------------------
__global__ __launch_bounds__(256)
void lstm_gemm_fused(const bf16_t* __restrict__ xcat,     // [B, Kc]
                     const bf16_t* __restrict__ wcat,     // [4*H, Kc]
                     const float*  __restrict__ cell,     // [B, H]
                     const float*  __restrict__ neighbors,// [B, 4]
                     const float*  __restrict__ Wn,       // [4, H, 4]
                     const float*  __restrict__ bias,     // [4, H]
                     float* __restrict__ out_hidden,      // [B, H]
                     float* __restrict__ out_cell) {      // [B, H]
    __shared__ __align__(16) bf16_t sA[2][BM][BK];           // 2 x 8 KB
    __shared__ __align__(16) bf16_t sB[2][Ngate][BN][BK];    // 2 x 8 KB

    const int t     = threadIdx.x;
    const int lane  = t & 31;
    const int wave  = t >> 5;          // 0..7
    const int waveM = wave & 3;        // 4 waves along M, 32 rows each
    const int waveN = wave >> 2;       // 2 waves along N, 16 cols each
    const int half  = lane >> 4;       // lane group 0/1
    const int l16   = lane & 15;

    const int m0 = blockIdx.y * BM;    // batch tile base
    const int h0 = blockIdx.x * BN;    // hidden tile base

    // --- global->LDS load assignments (all 16B b128 transactions) ---
    const int aRow = t >> 1;                 // 128 rows, 16 bf16 each (2 x b128)
    const int aCol = (t & 1) << 4;
    const int f0   = t << 4;                 // 16 bf16 of sB per thread
    const int bG   = f0 >> 10;               // gate
    const int bRow = (f0 & 1023) >> 5;
    const int bCol = f0 & 31;                // 0 or 16

    const bf16_t* aPtr = xcat + (size_t)(m0 + aRow) * Kc + aCol;
    const bf16_t* bPtr = wcat + (size_t)(bG * Hsz + h0 + bRow) * Kc + bCol;

    const int arBase = waveM * 32 + l16;     // A row of subtile 0
    const int br     = waveN * 16 + l16;     // B column (h) within tile

    v8f acc[2][Ngate] = {};

    // ---- prologue: stage K-step 0 into buffer 0 ----
#if USE_ASYNC_LDS
    async_b128(aPtr,     &sA[0][aRow][aCol]);
    async_b128(aPtr + 8, &sA[0][aRow][aCol + 8]);
    async_b128(bPtr,     &sB[0][bG][bRow][bCol]);
    async_b128(bPtr + 8, &sB[0][bG][bRow][bCol + 8]);
#else
    {
        bhalf8 ra0 = *(const bhalf8*)(aPtr);
        bhalf8 ra1 = *(const bhalf8*)(aPtr + 8);
        bhalf8 rb0 = *(const bhalf8*)(bPtr);
        bhalf8 rb1 = *(const bhalf8*)(bPtr + 8);
        *(bhalf8*)&sA[0][aRow][aCol]         = ra0;
        *(bhalf8*)&sA[0][aRow][aCol + 8]     = ra1;
        *(bhalf8*)&sB[0][bG][bRow][bCol]     = rb0;
        *(bhalf8*)&sB[0][bG][bRow][bCol + 8] = rb1;
    }
#endif

    for (int i = 0; i < NSTEP; ++i) {
        const int cur = i & 1;
#if USE_ASYNC_LDS
        wait_async0();
#endif
        __syncthreads();   // buf[cur] ready; everyone done reading buf[cur^1]

        // ---- prefetch next K-step into the other buffer ----
        if (i + 1 < NSTEP) {
            const int nxt = cur ^ 1;
            const int k0  = (i + 1) * BK;
#if USE_ASYNC_LDS
            async_b128(aPtr + k0,     &sA[nxt][aRow][aCol]);
            async_b128(aPtr + k0 + 8, &sA[nxt][aRow][aCol + 8]);
            async_b128(bPtr + k0,     &sB[nxt][bG][bRow][bCol]);
            async_b128(bPtr + k0 + 8, &sB[nxt][bG][bRow][bCol + 8]);
#else
            bhalf8 ra0 = *(const bhalf8*)(aPtr + k0);
            bhalf8 ra1 = *(const bhalf8*)(aPtr + k0 + 8);
            bhalf8 rb0 = *(const bhalf8*)(bPtr + k0);
            bhalf8 rb1 = *(const bhalf8*)(bPtr + k0 + 8);
            *(bhalf8*)&sA[nxt][aRow][aCol]         = ra0;
            *(bhalf8*)&sA[nxt][aRow][aCol + 8]     = ra1;
            *(bhalf8*)&sB[nxt][bG][bRow][bCol]     = rb0;
            *(bhalf8*)&sB[nxt][bG][bRow][bCol + 8] = rb1;
#endif
        }

        // ---- compute on buf[cur]: 2 M-subtiles x 4 gates = 8 WMMA ----
        // A fragment: lane holds K = {half*8+0..7, 16+half*8+0..7}
        v16bf af[2];
#pragma unroll
        for (int ms = 0; ms < 2; ++ms) {
            const int ar = arBase + ms * 16;
            bhalf8 a0 = *(const bhalf8*)&sA[cur][ar][half * 8];
            bhalf8 a1 = *(const bhalf8*)&sA[cur][ar][16 + half * 8];
            af[ms] = __builtin_shufflevector(a0, a1,
                        0,1,2,3,4,5,6,7,8,9,10,11,12,13,14,15);
        }
#pragma unroll
        for (int g = 0; g < Ngate; ++g) {
            // B fragment: lane n holds column n, K = {half*8+0..7, 16+half*8+0..7}
            bhalf8 b0 = *(const bhalf8*)&sB[cur][g][br][half * 8];
            bhalf8 b1 = *(const bhalf8*)&sB[cur][g][br][16 + half * 8];
            v16bf bfv = __builtin_shufflevector(b0, b1,
                          0,1,2,3,4,5,6,7,8,9,10,11,12,13,14,15);
            acc[0][g] = __builtin_amdgcn_wmma_f32_16x16x32_bf16(
                            false, af[0], false, bfv, (short)0, acc[0][g], false, false);
            acc[1][g] = __builtin_amdgcn_wmma_f32_16x16x32_bf16(
                            false, af[1], false, bfv, (short)0, acc[1][g], false, false);
        }
    }

    // ---------------- epilogue: neighbors, bias, activations ----------------
    const int gh = h0 + waveN * 16 + l16;     // hidden column of this lane

    float wn[Ngate][4];
    float bb[Ngate];
#pragma unroll
    for (int g = 0; g < Ngate; ++g) {
        float4 w = *(const float4*)(Wn + ((size_t)(g * Hsz + gh)) * 4);
        wn[g][0] = w.x; wn[g][1] = w.y; wn[g][2] = w.z; wn[g][3] = w.w;
        bb[g] = bias[g * Hsz + gh];
    }

#pragma unroll
    for (int ms = 0; ms < 2; ++ms) {
#pragma unroll
        for (int r = 0; r < 8; ++r) {
            const int gm = m0 + waveM * 32 + ms * 16 + half * 8 + r;  // C/D layout
            float4 nb = *(const float4*)(neighbors + (size_t)gm * 4);

            float pre[Ngate];
#pragma unroll
            for (int g = 0; g < Ngate; ++g) {
                pre[g] = acc[ms][g][r] + bb[g]
                       + nb.x * wn[g][0] + nb.y * wn[g][1]
                       + nb.z * wn[g][2] + nb.w * wn[g][3];
            }
            float ig = sigm(pre[0]);
            float fg = sigm(pre[1]);
            float og = sigm(pre[2]);
            float sg = tanhf(pre[3]);

            float cOld = cell[(size_t)gm * Hsz + gh];
            float cNew = fg * cOld + ig * sg;
            float hNew = og * tanhf(cNew);

            out_hidden[(size_t)gm * Hsz + gh] = hNew;
            out_cell  [(size_t)gm * Hsz + gh] = cNew;
        }
    }
}

// ---------------------------------------------------------------------------
extern "C" void kernel_launch(void* const* d_in, const int* in_sizes, int n_in,
                              void* d_out, int out_size, void* d_ws, size_t ws_size,
                              hipStream_t stream) {
    const float* x         = (const float*)d_in[0];
    const float* hidden    = (const float*)d_in[1];
    const float* cell      = (const float*)d_in[2];
    const float* neighbors = (const float*)d_in[3];
    const float* Wx        = (const float*)d_in[4];   // [4, H, IN]
    const float* Wh        = (const float*)d_in[5];   // [4, H, H]
    const float* Wn        = (const float*)d_in[6];   // [4, H, 4]
    const float* bias      = (const float*)d_in[7];   // [4, H]

    bf16_t* xcat = (bf16_t*)d_ws;                                        // 32 MB
    bf16_t* wcat = (bf16_t*)((char*)d_ws + (size_t)Bsz * Kc * 2);        // 16 MB

    float* out_hidden = (float*)d_out;
    float* out_cell   = out_hidden + (size_t)Bsz * Hsz;

    {   // pack activations: B*Kc/8 threads
        int nThreads = Bsz * Kc / 8;
        pack_xcat<<<nThreads / 256, 256, 0, stream>>>(x, hidden, xcat);
    }
    {   // pack weights: 4H*Kc/8 threads
        int nThreads = Ngate * Hsz * Kc / 8;
        pack_wcat<<<nThreads / 256, 256, 0, stream>>>(Wx, Wh, wcat);
    }

    dim3 grid(Hsz / BN, Bsz / BM);   // 32 x 64 workgroups
    lstm_gemm_fused<<<grid, 256, 0, stream>>>(xcat, wcat, cell, neighbors,
                                              Wn, bias, out_hidden, out_cell);
}
